// QLoRALinear_46557445489443
// MI455X (gfx1250) — matrix operand order repaired and checked by reading, not compiled, not run on hardware
//
#include <hip/hip_runtime.h>
#include <hip/hip_bf16.h>

// ---------------------------------------------------------------------------
// QLoRA linear on MI455X (gfx1250):
//   out[8192,4096] = x[8192,4096] . W'[4096,4096]^T
// where W' = scale*Wq + 0.25 * lora_B . lora_A  (rank-64 update folded into
// the weight once, in bf16, into d_ws).  Main GEMM uses wave32 WMMA
// v_wmma_f32_16x16x32_bf16 with fp32 accumulation.
// Block tile 128x256 (8 waves, 2x4), wave tile 64x64 -> 16 WMMAs / K-step
// per wave to maximize WMMA density per LDS byte.
// ---------------------------------------------------------------------------

typedef __attribute__((ext_vector_type(16))) __bf16 v16bf;
typedef __attribute__((ext_vector_type(8)))  float  v8f;

#define K_DIM 4096
#define N_DIM 4096
#define M_DIM 8192

#define BM 128
#define BN 256
#define BK 32
#define LDT 40   // LDS row stride in bf16 elems (BK + 8 pad), keeps 16B align

// ---------------------------------------------------------------------------
// Kernel 1: fold dequant + LoRA into a bf16 weight matrix W'[n][k] in d_ws.
//   W'[n,k] = scale * Wq[n,k] + 0.25 * sum_r lora_B[n,r] * lora_A[r,k]
// 2.1 GFLOP total; lora_A (1MB) and lora_B (1MB) live in L2; k-adjacent
// threads give coalesced lora_A reads and broadcast lora_B reads.
// ---------------------------------------------------------------------------
__global__ __launch_bounds__(256) void qlora_prep_weight(
    const int* __restrict__ wq,        // [N,K] int (values in [-8,7])
    const float* __restrict__ wscale,  // [1]
    const float* __restrict__ lA,      // [64, K]
    const float* __restrict__ lB,      // [N, 64]
    unsigned short* __restrict__ wp)   // [N,K] bf16 bits (workspace)
{
    const int idx = blockIdx.x * 256 + threadIdx.x;   // over N*K
    const int n = idx >> 12;          // /4096
    const int k = idx & (K_DIM - 1);

    const float* bRow = lB + n * 64;
    float acc = 0.0f;
#pragma unroll 4
    for (int r = 0; r < 64; ++r)
        acc = fmaf(bRow[r], lA[r * K_DIM + k], acc);

    float v = fmaf((float)wq[idx], wscale[0], 0.25f * acc);
    __bf16 h = (__bf16)v;
    wp[idx] = __builtin_bit_cast(unsigned short, h);
}

// ---------------------------------------------------------------------------
// Kernel 2: bf16 WMMA GEMM.  Block = 256 threads = 8 waves, 128x256 C tile.
// Waves laid out 2 (M) x 4 (N); each wave owns 4x4 = 16 WMMA 16x16 tiles.
// x is converted fp32->bf16 on the fly while staging into LDS.
// ---------------------------------------------------------------------------
__global__ __launch_bounds__(256) void qlora_gemm_bf16_wmma(
    const float* __restrict__ x,            // [M,K] fp32
    const unsigned short* __restrict__ wpb, // [N,K] bf16 bits
    float* __restrict__ out)                // [M,N] fp32
{
    __shared__ __bf16 sA[BM * LDT];   // 10240 B
    __shared__ __bf16 sB[BN * LDT];   // 20480 B

    const __bf16* wp = (const __bf16*)wpb;

    const int t    = threadIdx.x;
    const int lane = t & 31;
    const int wid  = t >> 5;          // 0..7
    const int wm   = wid >> 2;        // 0..1  -> 64 M rows each
    const int wn   = wid & 3;         // 0..3  -> 64 N cols each

    const int mBase = blockIdx.y * BM;
    const int nBase = blockIdx.x * BN;

    // A staging: 2 threads per row, 16 fp32 elements each
    const int aRow = t >> 1;          // 0..127
    const int aSeg = (t & 1) * 16;    // 0 or 16
    const float*  xRow = x  + (size_t)(mBase + aRow) * K_DIM + aSeg;
    // B staging: 1 thread per row, full 32-bf16 (64 B) segment
    const __bf16* wRow = wp + (size_t)(nBase + t) * K_DIM;

    v8f acc[4][4] = {};               // fp32 accumulators (128 VGPRs)

    const int lrow = lane & 15;       // M (A frag) / N (B frag) within tile
    const int kh   = lane >> 4;       // K-half selector: k = 16*kh + 0..15 (contiguous order)

    for (int k0 = 0; k0 < K_DIM; k0 += BK) {
        // ---- stage A: 16 fp32 -> 16 bf16 -> LDS (32 bytes) ----
        const float4* xs = (const float4*)(xRow + k0);
        float4 f0 = xs[0], f1 = xs[1], f2 = xs[2], f3 = xs[3];
        v16bf av;
        av[0]  = (__bf16)f0.x; av[1]  = (__bf16)f0.y;
        av[2]  = (__bf16)f0.z; av[3]  = (__bf16)f0.w;
        av[4]  = (__bf16)f1.x; av[5]  = (__bf16)f1.y;
        av[6]  = (__bf16)f1.z; av[7]  = (__bf16)f1.w;
        av[8]  = (__bf16)f2.x; av[9]  = (__bf16)f2.y;
        av[10] = (__bf16)f2.z; av[11] = (__bf16)f2.w;
        av[12] = (__bf16)f3.x; av[13] = (__bf16)f3.y;
        av[14] = (__bf16)f3.z; av[15] = (__bf16)f3.w;
        {
            uint4* d = (uint4*)&sA[aRow * LDT + aSeg];
            const uint4* s = (const uint4*)&av;
            d[0] = s[0]; d[1] = s[1];
        }
        // ---- stage B: straight 64-byte bf16 row copy -> LDS ----
        {
            const uint4* s = (const uint4*)(wRow + k0);
            uint4* d = (uint4*)&sB[t * LDT];
            d[0] = s[0]; d[1] = s[1]; d[2] = s[2]; d[3] = s[3];
        }
        if (k0 + BK < K_DIM) {
            __builtin_prefetch(wRow + k0 + BK, 0, 1);   // global_prefetch_b8
            __builtin_prefetch(xRow + k0 + BK, 0, 1);
        }
        __syncthreads();

        // ---- B fragments (resident): 32x16 bf16, lane%16 = N column,
        //      K half = 16*(lane>=16), 16 contiguous bf16 per lane ----
        v16bf bfrag[4];
#pragma unroll
        for (int nt = 0; nt < 4; ++nt) {
            const __bf16* pb = &sB[(wn * 64 + nt * 16 + lrow) * LDT + kh * 16];
            uint4* d = (uint4*)&bfrag[nt];
            d[0] = *(const uint4*)pb;
            d[1] = *(const uint4*)(pb + 8);
        }

        // ---- stream A fragments one m-tile at a time: 16 WMMAs / K-step ----
        // A 16x32 bf16: K = 16*(lane>=16) + 0..15, contiguous per lane
#pragma unroll
        for (int mt = 0; mt < 4; ++mt) {
            const __bf16* pa = &sA[(wm * 64 + mt * 16 + lrow) * LDT + kh * 16];
            v16bf afrag;
            {
                uint4* d = (uint4*)&afrag;
                d[0] = *(const uint4*)pa;
                d[1] = *(const uint4*)(pa + 8);
            }
#pragma unroll
            for (int nt = 0; nt < 4; ++nt)
                acc[mt][nt] = __builtin_amdgcn_wmma_f32_16x16x32_bf16(
                    false, afrag, false, bfrag[nt],
                    (short)0, acc[mt][nt], false, false);
        }

        __syncthreads();
    }

    // ---- epilogue: C/D layout -> global (VGPR j: M=j lanes 0-15, M=j+8 hi)
    const int rowOff = (lane >> 4) * 8;
#pragma unroll
    for (int mt = 0; mt < 4; ++mt) {
#pragma unroll
        for (int nt = 0; nt < 4; ++nt) {
            const int col = nBase + wn * 64 + nt * 16 + lrow;
#pragma unroll
            for (int j = 0; j < 8; ++j) {
                const int row = mBase + wm * 64 + mt * 16 + rowOff + j;
                out[(size_t)row * N_DIM + col] = acc[mt][nt][j];
            }
        }
    }
}

// ---------------------------------------------------------------------------
extern "C" void kernel_launch(void* const* d_in, const int* in_sizes, int n_in,
                              void* d_out, int out_size, void* d_ws, size_t ws_size,
                              hipStream_t stream) {
    const float* x      = (const float*)d_in[0];  // [4,2048,4096]
    const int*   wq     = (const int*)d_in[1];    // [4096,4096]
    const float* wscale = (const float*)d_in[2];  // [1]
    const float* lA     = (const float*)d_in[3];  // [64,4096]
    const float* lB     = (const float*)d_in[4];  // [4096,64]
    float* out = (float*)d_out;
    unsigned short* wp = (unsigned short*)d_ws;   // 4096*4096 bf16 = 32 MB

    // 1) fold dequant + rank-64 LoRA into bf16 W'
    qlora_prep_weight<<<(N_DIM * K_DIM) / 256, 256, 0, stream>>>(
        wq, wscale, lA, lB, wp);

    // 2) one big bf16 WMMA GEMM: 8192 x 4096 x 4096
    dim3 grid(N_DIM / BN, M_DIM / BM);  // (16, 64)
    qlora_gemm_bf16_wmma<<<grid, 256, 0, stream>>>(x, wp, out);
}
